// PPOLSTMModel_70153995813047
// MI455X (gfx1250) — matrix operand order repaired and checked by reading, not compiled, not run on hardware
//
#include <hip/hip_runtime.h>
#include <hip/hip_bf16.h>
#include <math.h>

// ---------------- problem constants ----------------
#define TB_TOT 65536   // T*B rows
#define BATCH  256     // B
#define TSTEPS 256     // T
#define LH     512     // L == D
#define NACT   1000    // num actions
#define NACT_P 1008    // padded to 63 WMMA tiles
#define SCAN_BLOCKS  32
#define SCAN_THREADS 512
#define LDSPITCH 520   // 520 halfs/row = 1040 B -> 4-bank advance, conflict-free

typedef _Float16 h16;
typedef __attribute__((ext_vector_type(8)))  _Float16 v8h;
typedef __attribute__((ext_vector_type(16))) _Float16 v16h;
typedef __attribute__((ext_vector_type(8)))  float    v8f;

union V16U { v16h v; v8h h[2]; };

__device__ __forceinline__ v8f wmma16(v16h a, v16h b, v8f c) {
  // D = A(16x32 f16) x B(32x16 f16) + C(16x16 f32)
  return __builtin_amdgcn_wmma_f32_16x16x32_f16(false, a, false, b, (short)0, c,
                                                false, false);
}

// A fragment (16 n x 32 k) from row-major f16, arbitrary ld (global or LDS).
__device__ __forceinline__ v16h frag_a(const h16* __restrict__ base, int ld, int k) {
  int lane = threadIdx.x & 31;
  const h16* p = base + (size_t)(lane & 15) * ld + k + (lane >> 4) * 8;
  V16U u;
  u.h[0] = *(const v8h*)(p);
  u.h[1] = *(const v8h*)(p + 16);
  return u.v;
}
// B fragment: W row-major (M x K); B = W^T tile (32k x 16m).
__device__ __forceinline__ v16h frag_b(const h16* __restrict__ W, int ld, int mBase, int k) {
  int lane = threadIdx.x & 31;
  const h16* p = W + (size_t)(mBase + (lane & 15)) * ld + k + (lane >> 4) * 8;
  V16U u;
  u.h[0] = *(const v8h*)(p);
  u.h[1] = *(const v8h*)(p + 16);
  return u.v;
}

__device__ __forceinline__ float sigf(float x) { return 1.0f / (1.0f + expf(-x)); }

// ---------------- f32 -> f16 convert (with zero padding) ----------------
__global__ void __launch_bounds__(256) cvt16(const float* __restrict__ src,
                                             h16* __restrict__ dst,
                                             long long n, long long total) {
  long long i = (long long)blockIdx.x * 256 + threadIdx.x;
  if (i < total) dst[i] = (i < n) ? (h16)src[i] : (h16)0.0f;
}

// ---------------- Y = tanh(X @ W^T + bias), N x 512 x 512 ----------------
// block: 32 rows (A tile staged in LDS once, shared by all 8 waves),
// wave j owns a 64-wide m-quad. grid.x = N/32.
__global__ void __launch_bounds__(256) gemm512(const h16* __restrict__ X,
                                               const h16* __restrict__ W,
                                               const float* __restrict__ bias,
                                               h16* __restrict__ Y) {
  __shared__ h16 alds[32 * LDSPITCH];        // 33,280 B
  const int nBase = blockIdx.x << 5;

  {  // cooperative A-tile stage: 32 rows x 512 halfs, 8 threads/row
    int r = threadIdx.x >> 3;
    int sub = threadIdx.x & 7;               // 64 halfs each
    const uint4* sp = (const uint4*)(X + (size_t)(nBase + r) * LH + sub * 64);
    uint4* dp = (uint4*)(&alds[r * LDSPITCH + sub * 64]);
#pragma unroll
    for (int q = 0; q < 8; ++q) dp[q] = sp[q];
  }
  __syncthreads();

  const int mBase = (threadIdx.x >> 5) << 6;   // wave -> m-quad
  v8f acc[2][4] = {};
  for (int k = 0; k < LH; k += 32) {
    v16h a0 = frag_a(&alds[0], LDSPITCH, k);
    v16h a1 = frag_a(&alds[16 * LDSPITCH], LDSPITCH, k);
#pragma unroll
    for (int j = 0; j < 4; ++j) {
      v16h b = frag_b(W, LH, mBase + j * 16, k);
      acc[0][j] = wmma16(a0, b, acc[0][j]);
      acc[1][j] = wmma16(a1, b, acc[1][j]);
    }
  }
  int lane = threadIdx.x & 31, mloc = lane & 15, hi = lane >> 4;
#pragma unroll
  for (int j = 0; j < 4; ++j) {
    int m = mBase + j * 16 + mloc;
    float bv = bias[m];
#pragma unroll
    for (int half = 0; half < 2; ++half) {
#pragma unroll
      for (int r = 0; r < 8; ++r) {
        int n = nBase + half * 16 + r + 8 * hi;
        Y[(size_t)n * LH + m] = (h16)tanhf(acc[half][j][r] + bv);
      }
    }
  }
}

// ---------------- LSTM state init ----------------
__global__ void __launch_bounds__(256) init_state(const float* __restrict__ h0,
                                                  const float* __restrict__ c0,
                                                  h16* __restrict__ hping,
                                                  float* __restrict__ cbuf,
                                                  unsigned* __restrict__ syncc) {
  int i = blockIdx.x * 256 + threadIdx.x;
  if (i < BATCH * LH) {
    hping[i] = (h16)h0[i];
    cbuf[i]  = c0[i];
  }
  if (blockIdx.x == 0 && threadIdx.x == 0) *syncc = 0u;
}

// ---------------- persistent LSTM scan ----------------
// 32 blocks x 16 waves = 512 waves = one per 16(batch) x 16(jj) tile.
// Block owns one jj-tile; Wih/Whh rows for its 64 gate columns staged in LDS
// once (conflict-free ds_load_b128 fragments). Each wave computes matching
// i,f,g,o tiles so the cell update is register-local. Grid barrier per step
// (monotonic counter) + agent-scope acquire fence so h-state is never stale.
__global__ void __launch_bounds__(SCAN_THREADS, 1) lstm_scan(
    const h16* __restrict__ h2,      // (T*B, L) f16 trunk output
    const int* __restrict__ done,    // (T*B,)
    const h16* __restrict__ Wih16,   // (4L, L) f16
    const h16* __restrict__ Whh16,   // (4L, L) f16
    const float* __restrict__ bih, const float* __restrict__ bhh,
    h16* __restrict__ hbuf,          // 2 * B*L f16 ping-pong h state
    float* __restrict__ cbuf,        // B*L f32 c state
    h16* __restrict__ NH,            // (T*B, L) f16 output (residual added)
    unsigned* __restrict__ syncc) {
  __shared__ h16 wlds[2 * 64 * LDSPITCH];   // 133,120 B

  const int tid = threadIdx.x;
  const int lane = tid & 31, waveInBlk = tid >> 5;   // 0..15
  const int mloc = lane & 15, hi = lane >> 4;
  const int nBase = waveInBlk * 16;                  // batch tile
  const int jjBase = (int)blockIdx.x * 16;           // hidden-column tile
  const int jj = jjBase + mloc;

  {  // stage Wih (rows 0..63) + Whh (rows 64..127): 4 threads per row
    int row = tid >> 2;          // 0..127
    int sub = tid & 3;           // 128 halfs each
    const h16* src = (row < 64) ? Wih16 : Whh16;
    int mr = row & 63;
    int grow = ((mr >> 4) * 512) + jjBase + (mr & 15);   // gate*512 + column
    const uint4* sp = (const uint4*)(src + (size_t)grow * LH + sub * 128);
    uint4* dp = (uint4*)(&wlds[row * LDSPITCH + sub * 128]);
#pragma unroll
    for (int q = 0; q < 16; ++q) dp[q] = sp[q];
  }
  __syncthreads();

  float bias4[4];
#pragma unroll
  for (int g = 0; g < 4; ++g) bias4[g] = bih[g * 512 + jj] + bhh[g * 512 + jj];

  for (int t = 0; t < TSTEPS; ++t) {
    const h16* hprev = hbuf + (size_t)(t & 1) * (BATCH * LH);
    h16* hnext = hbuf + (size_t)((t + 1) & 1) * (BATCH * LH);
    const int* dn = done + t * BATCH;
    h16 msk = dn[nBase + mloc] ? (h16)0.0f : (h16)1.0f;   // hidden reset mask

    v8f acc[4] = {};
#pragma unroll
    for (int half = 0; half < 2; ++half) {
      const h16* abase = (half == 0)
          ? (h2 + ((size_t)t * BATCH + nBase) * LH)     // x-part (Wih)
          : (hprev + (size_t)nBase * LH);               // h-part (Whh)
      const h16* wbase = &wlds[half * (64 * LDSPITCH)];
      for (int kc = 0; kc < 16; ++kc) {
        V16U ua;
        const h16* p = abase + (size_t)mloc * LH + kc * 32 + hi * 8;
        ua.h[0] = *(const v8h*)(p);
        ua.h[1] = *(const v8h*)(p + 16);
        if (half == 0) {
          // prefetch next timestep's trunk row (global_prefetch_b8)
          if (t + 1 < TSTEPS) __builtin_prefetch(p + (size_t)BATCH * LH, 0, 3);
        } else {
          ua.h[0] *= msk;
          ua.h[1] *= msk;
        }
#pragma unroll
        for (int g = 0; g < 4; ++g) {
          const h16* wp = wbase + (g * 16 + mloc) * LDSPITCH + kc * 32 + hi * 8;
          V16U ub;
          ub.h[0] = *(const v8h*)(wp);
          ub.h[1] = *(const v8h*)(wp + 16);
          acc[g] = wmma16(ua.v, ub.v, acc[g]);
        }
      }
    }

    // cell update + residual, register-local
#pragma unroll
    for (int r = 0; r < 8; ++r) {
      int b = nBase + r + 8 * hi;
      float iv = acc[0][r] + bias4[0];
      float fv = acc[1][r] + bias4[1];
      float gv = acc[2][r] + bias4[2];
      float ov = acc[3][r] + bias4[3];
      float cp = dn[b] ? 0.0f : cbuf[b * LH + jj];
      float cn = sigf(fv) * cp + sigf(iv) * tanhf(gv);
      float hn = sigf(ov) * tanhf(cn);
      cbuf[b * LH + jj] = cn;
      hnext[b * LH + jj] = (h16)hn;
      size_t orow = ((size_t)t * BATCH + b) * LH + jj;
      NH[orow] = (h16)(hn + (float)h2[orow]);
    }

    // ---- grid barrier: release stores, arrive, spin, acquire ----
    __threadfence();
    __syncthreads();
    if (tid == 0) {
      __hip_atomic_fetch_add(syncc, 1u, __ATOMIC_RELEASE, __HIP_MEMORY_SCOPE_AGENT);
      unsigned target = (unsigned)(t + 1) * SCAN_BLOCKS;
      while (__hip_atomic_load(syncc, __ATOMIC_ACQUIRE, __HIP_MEMORY_SCOPE_AGENT) < target)
        __builtin_amdgcn_s_sleep(8);
    }
    __syncthreads();
    // invalidate vector caches so hprev reads are never stale (LDS weights untouched)
    __builtin_amdgcn_fence(__ATOMIC_ACQUIRE, "agent");
  }
}

// ---------------- fused actor logits + log-softmax + gather/entropy --------
// block = 16 rows; A tile staged in LDS (reused by 63 column tiles);
// logits live only in LDS; half-wave shuffle reductions.
__global__ void __launch_bounds__(256) actor_logits(const h16* __restrict__ Aact,
                                                    const h16* __restrict__ Wa2,
                                                    const float* __restrict__ ba2,
                                                    const int* __restrict__ action,
                                                    float* __restrict__ out_lp,
                                                    float* __restrict__ out_ent) {
  __shared__ float slog[16 * NACT];          // 64,000 B
  __shared__ h16 alds[16 * LDSPITCH];        // 16,640 B
  const int nBase = blockIdx.x << 4;
  const int tid = threadIdx.x, lane = tid & 31, wave = tid >> 5;
  const int mloc = lane & 15, hi = lane >> 4;

  {  // stage 16-row A tile: 16 threads/row, 32 halfs each
    int r = tid >> 4;
    int sub = tid & 15;
    const uint4* sp = (const uint4*)(Aact + (size_t)(nBase + r) * LH + sub * 32);
    uint4* dp = (uint4*)(&alds[r * LDSPITCH + sub * 32]);
#pragma unroll
    for (int q = 0; q < 4; ++q) dp[q] = sp[q];
  }
  __syncthreads();

  for (int ct = wave; ct < NACT_P / 16; ct += 8) {   // wave-uniform trip count
    int mBase = ct << 4;
    v8f acc = {};
    for (int k = 0; k < LH; k += 32) {
      v16h a = frag_a(&alds[0], LDSPITCH, k);
      v16h b = frag_b(Wa2, LH, mBase, k);
      acc = wmma16(a, b, acc);
    }
    int m = mBase + mloc;
    if (m < NACT) {
      float bv = ba2[m];
#pragma unroll
      for (int r = 0; r < 8; ++r) slog[(r + 8 * hi) * NACT + m] = acc[r] + bv;
    }
  }
  __syncthreads();

  int row = tid >> 4, sub = tid & 15;     // 16 threads per row
  float pmax = -3.4e38f;
  for (int c = sub; c < NACT; c += 16) pmax = fmaxf(pmax, slog[row * NACT + c]);
#pragma unroll
  for (int o = 8; o; o >>= 1) pmax = fmaxf(pmax, __shfl_xor(pmax, o, 16));
  float z = 0.0f, s1 = 0.0f;
  for (int c = sub; c < NACT; c += 16) {
    float s = slog[row * NACT + c] - pmax;
    float e = expf(s);
    z += e;
    s1 += e * s;
  }
#pragma unroll
  for (int o = 8; o; o >>= 1) {
    z  += __shfl_xor(z, o, 16);
    s1 += __shfl_xor(s1, o, 16);
  }
  if (sub == 0) {
    int n = nBase + row;
    float logz = logf(z);
    float sa = slog[row * NACT + action[n]] - pmax;
    out_lp[n]  = sa - logz;           // log pi(a|s)
    out_ent[n] = logz - s1 / z;       // entropy
  }
}

// ---------------- critic value: v = ch . Wc2 + bc2 (wave per row) ----------
__global__ void __launch_bounds__(256) critic_v(const h16* __restrict__ CH,
                                                const float* __restrict__ Wc2,
                                                const float* __restrict__ bc2,
                                                float* __restrict__ out_v, int N) {
  int gw = (int)((blockIdx.x * 256u + threadIdx.x) >> 5);
  if (gw >= N) return;
  int lane = threadIdx.x & 31;
  const h16* row = CH + (size_t)gw * LH;
  float s = 0.0f;
  int k0 = lane * 16;
#pragma unroll
  for (int k = 0; k < 16; ++k) s += (float)row[k0 + k] * Wc2[k0 + k];
#pragma unroll
  for (int o = 16; o; o >>= 1) s += __shfl_xor(s, o, 32);
  if (lane == 0) out_v[gw] = s + bc2[0];
}

// ---------------- host-side orchestration ----------------
extern "C" void kernel_launch(void* const* d_in, const int* in_sizes, int n_in,
                              void* d_out, int out_size, void* d_ws, size_t ws_size,
                              hipStream_t stream) {
  (void)in_sizes; (void)n_in; (void)out_size; (void)ws_size;
  const float* x     = (const float*)d_in[0];
  const int*   done  = (const int*)d_in[1];
  const int*   action= (const int*)d_in[2];
  // d_in[3] = B (scalar, known == 256)
  const float* W_t1  = (const float*)d_in[4];
  const float* b_t1  = (const float*)d_in[5];
  const float* W_t2  = (const float*)d_in[6];
  const float* b_t2  = (const float*)d_in[7];
  const float* Wih   = (const float*)d_in[8];
  const float* Whh   = (const float*)d_in[9];
  const float* bih   = (const float*)d_in[10];
  const float* bhh   = (const float*)d_in[11];
  const float* Wa1   = (const float*)d_in[12];
  const float* ba1   = (const float*)d_in[13];
  const float* Wa2   = (const float*)d_in[14];
  const float* ba2   = (const float*)d_in[15];
  const float* Wc1   = (const float*)d_in[16];
  const float* bc1   = (const float*)d_in[17];
  const float* Wc2   = (const float*)d_in[18];
  const float* bc2   = (const float*)d_in[19];
  const float* h0    = (const float*)d_in[20];
  const float* c0    = (const float*)d_in[21];
  float* out = (float*)d_out;

  // ---- carve workspace (256-byte aligned regions) ----
  size_t off = 0;
  char* base = (char*)d_ws;
  auto carve = [&](size_t bytes) -> void* {
    void* p = base + off;
    off += (bytes + 255) & ~(size_t)255;
    return p;
  };
  h16* x16   = (h16*)carve((size_t)TB_TOT * LH * 2);   // x f16, reused as NH
  h16* bufA  = (h16*)carve((size_t)TB_TOT * LH * 2);   // h1 -> a -> ch
  h16* bufB  = (h16*)carve((size_t)TB_TOT * LH * 2);   // h2 (live through scan)
  h16* wt1   = (h16*)carve((size_t)512 * 512 * 2);
  h16* wt2   = (h16*)carve((size_t)512 * 512 * 2);
  h16* wih16 = (h16*)carve((size_t)2048 * 512 * 2);
  h16* whh16 = (h16*)carve((size_t)2048 * 512 * 2);
  h16* wa116 = (h16*)carve((size_t)512 * 512 * 2);
  h16* wa216 = (h16*)carve((size_t)NACT_P * 512 * 2);  // zero padded rows
  h16* wc116 = (h16*)carve((size_t)512 * 512 * 2);
  float* cbuf = (float*)carve((size_t)BATCH * LH * 4);
  h16*  hbuf  = (h16*)carve((size_t)2 * BATCH * LH * 2);
  unsigned* syncc = (unsigned*)carve(256);

  auto nb = [](long long n) { return (unsigned)((n + 255) / 256); };

  // ---- f32 -> f16 conversions ----
  long long nx = (long long)TB_TOT * LH;
  cvt16<<<nb(nx), 256, 0, stream>>>(x, x16, nx, nx);
  cvt16<<<nb(512 * 512), 256, 0, stream>>>(W_t1, wt1, 512 * 512, 512 * 512);
  cvt16<<<nb(512 * 512), 256, 0, stream>>>(W_t2, wt2, 512 * 512, 512 * 512);
  cvt16<<<nb(2048 * 512), 256, 0, stream>>>(Wih, wih16, 2048 * 512, 2048 * 512);
  cvt16<<<nb(2048 * 512), 256, 0, stream>>>(Whh, whh16, 2048 * 512, 2048 * 512);
  cvt16<<<nb(512 * 512), 256, 0, stream>>>(Wa1, wa116, 512 * 512, 512 * 512);
  cvt16<<<nb((long long)NACT_P * 512), 256, 0, stream>>>(Wa2, wa216,
      (long long)NACT * 512, (long long)NACT_P * 512);
  cvt16<<<nb(512 * 512), 256, 0, stream>>>(Wc1, wc116, 512 * 512, 512 * 512);

  // ---- trunk ----
  gemm512<<<TB_TOT / 32, 256, 0, stream>>>(x16, wt1, b_t1, bufA);
  gemm512<<<TB_TOT / 32, 256, 0, stream>>>(bufA, wt2, b_t2, bufB);

  // ---- LSTM scan (persistent, grid-synced) ----
  init_state<<<nb(BATCH * LH), 256, 0, stream>>>(h0, c0, hbuf, cbuf, syncc);
  lstm_scan<<<SCAN_BLOCKS, SCAN_THREADS, 0, stream>>>(bufB, done, wih16, whh16,
                                                      bih, bhh, hbuf, cbuf,
                                                      x16 /*NH*/, syncc);

  // ---- actor head ----
  gemm512<<<TB_TOT / 32, 256, 0, stream>>>(x16, wa116, ba1, bufA);
  actor_logits<<<TB_TOT / 16, 256, 0, stream>>>(bufA, wa216, ba2, action,
                                                out, out + TB_TOT);

  // ---- critic head ----
  gemm512<<<TB_TOT / 32, 256, 0, stream>>>(x16, wc116, bc1, bufA);
  critic_v<<<TB_TOT / 8, 256, 0, stream>>>(bufA, Wc2, bc2, out + 2 * TB_TOT, TB_TOT);
}
